// StudentDecoderAttention_4758823764609
// MI455X (gfx1250) — matrix operand order repaired and checked by reading, not compiled, not run on hardware
//
#include <hip/hip_runtime.h>
#include <hip/hip_bf16.h>
#include <math.h>

// ---------------------------------------------------------------------------
// StudentDecoderAttention on MI455X (gfx1250, wave32, WMMA).
// Pipeline: gen weights (f16) -> cast x (f16) -> QKV GEMM (wmma f16->f32,
//           32x64 wave tile) -> flash attention (wmma + online softmax)
//           -> proj GEMM (32x64 wave tile) -> f32 out
// ---------------------------------------------------------------------------

typedef __attribute__((ext_vector_type(16))) _Float16 v16h;
typedef __attribute__((ext_vector_type(8)))  _Float16 v8h;
typedef __attribute__((ext_vector_type(8)))  float    v8f;

#define B_  8
#define S_  1024
#define E_  1024
#define H_  16
#define D_  64

// ---- WMMA fragment loaders (contiguous k order, same in A and B, wave32) ----------
// A 16x32 f16: lane<16 -> row=lane, K = k0+0..15
//              lane>=16 -> row=lane-16, K = k0+16..31
__device__ __forceinline__ v16h ld_frag_a(const _Float16* base, int ld, int lane) {
    const _Float16* p = base + (lane & 15) * ld + ((lane >> 4) << 4);
    return *(const v16h*)p;
}

// B 32x16 f16 from a row-major matrix M[n, k] (B[k,n] = M[n0+n, k0+k]):
// lane<16 -> col=lane, K=k0+0..15 ; lane>=16 -> col=lane-16, K=k0+16..31
// => one contiguous 32-byte load per lane.
__device__ __forceinline__ v16h ld_frag_b(const _Float16* base, int ld, int lane) {
    const _Float16* p = base + (lane & 15) * ld + ((lane >> 4) << 4);
    return *(const v16h*)p;
}

__device__ __forceinline__ v8f wmma_f16(v16h a, v16h b, v8f c) {
    return __builtin_amdgcn_wmma_f32_16x16x32_f16(false, a, false, b,
                                                  (short)0, c, false, false);
}

// ---- Weight / bias generators ---------------------------------------------
__global__ void __launch_bounds__(256)
gen_weight_f16(const float* __restrict__ sub, const float* __restrict__ wl,
               const float* __restrict__ W, const float* __restrict__ Bp,
               _Float16* __restrict__ out, int n) {
    int i = blockIdx.x * blockDim.x + threadIdx.x;
    if (i >= n) return;
    float base = wl[0] * sub[i] + wl[1] * sub[i + n];   // L_ADJ == 2
    out[i] = (_Float16)(tanhf(base) * W[i] + Bp[i]);
}

__global__ void __launch_bounds__(256)
gen_bias_f32(const float* __restrict__ sub, const float* __restrict__ wl,
             const float* __restrict__ W, const float* __restrict__ Bp,
             float* __restrict__ out, int n) {
    int i = blockIdx.x * blockDim.x + threadIdx.x;
    if (i >= n) return;
    float base = wl[0] * sub[i] + wl[1] * sub[i + n];
    out[i] = tanhf(base) * W[i] + Bp[i];
}

__global__ void __launch_bounds__(256)
cast_to_f16(const float* __restrict__ in, _Float16* __restrict__ out) {
    int i = (blockIdx.x * blockDim.x + threadIdx.x) << 2;
    float4 v = *(const float4*)(in + i);
    out[i + 0] = (_Float16)v.x;
    out[i + 1] = (_Float16)v.y;
    out[i + 2] = (_Float16)v.z;
    out[i + 3] = (_Float16)v.w;
}

// ---- QKV GEMM: [B*S, E] x [3E, E]^T + bias -> Q,K [B,H,S,D], Vt [B,H,D,S] --
// Wave tile 32x64: 2 A-fragments share 4 B-fragments -> 8 WMMAs per K-step.
__global__ void __launch_bounds__(256)
gemm_qkv(const _Float16* __restrict__ X, const _Float16* __restrict__ W,
         const float* __restrict__ bias,
         _Float16* __restrict__ qb, _Float16* __restrict__ kb,
         _Float16* __restrict__ vtb) {
    const int lane = threadIdx.x & 31;
    const int gw   = blockIdx.x * 8 + (threadIdx.x >> 5);
    const int TN   = (3 * E_) / 64;                 // 48 n-tiles
    const int m0   = (gw / TN) << 5;                // 32-row tiles
    const int n0   = (gw % TN) << 6;

    v8f accLo[4] = {}, accHi[4] = {};
    const _Float16* xlo = X + (size_t)m0 * E_;
    const _Float16* xhi = X + (size_t)(m0 + 16) * E_;
    for (int k0 = 0; k0 < E_; k0 += 32) {
        __builtin_prefetch(xlo + k0 + 256, 0, 0);   // global_prefetch_b8
        const v16h aLo = ld_frag_a(xlo + k0, E_, lane);
        const v16h aHi = ld_frag_a(xhi + k0, E_, lane);
#pragma unroll
        for (int t = 0; t < 4; ++t) {
            const v16h bf = ld_frag_b(W + (size_t)(n0 + (t << 4)) * E_ + k0,
                                      E_, lane);
            accLo[t] = wmma_f16(aLo, bf, accLo[t]);
            accHi[t] = wmma_f16(aHi, bf, accHi[t]);
        }
    }

    const int row = (lane >> 4) << 3;   // C/D layout: row = r + 8*(lane>=16)
    const int col = lane & 15;
#pragma unroll
    for (int t = 0; t < 4; ++t) {
        const int oc   = n0 + (t << 4) + col;
        const float bv = bias[oc];
        const int sect = oc >> 10;          // 0=q 1=k 2=v
        const int rem  = oc & 1023;
        const int h    = rem >> 6;
        const int d    = rem & 63;
#pragma unroll
        for (int half = 0; half < 2; ++half) {
            const v8f acc = half ? accHi[t] : accLo[t];
#pragma unroll
            for (int r = 0; r < 8; ++r) {
                const int m  = m0 + (half << 4) + row + r;
                const int bb = m >> 10;         // S_ == 1024
                const int ss = m & 1023;
                const _Float16 hv = (_Float16)(acc[r] + bv);
                if (sect == 0)
                    qb [(((size_t)(bb * H_ + h)) * S_ + ss) * D_ + d] = hv;
                else if (sect == 1)
                    kb [(((size_t)(bb * H_ + h)) * S_ + ss) * D_ + d] = hv;
                else
                    vtb[(((size_t)(bb * H_ + h)) * D_ + d) * S_ + ss] = hv;
            }
        }
    }
}

// ---- Flash attention: one wave per 16-query tile ---------------------------
__global__ void __launch_bounds__(256)
attn_kernel(const _Float16* __restrict__ qb, const _Float16* __restrict__ kb,
            const _Float16* __restrict__ vtb, _Float16* __restrict__ attn_out) {
    __shared__ __align__(128) _Float16 lds_p[8][16 * 32];  // per-wave P staging

    const int lane = threadIdx.x & 31;
    const int wid  = threadIdx.x >> 5;
    const int tile = blockIdx.x * 8 + wid;     // 0 .. B*H*(S/16)-1
    const int qt   = tile & (S_ / 16 - 1);
    const int bh   = tile >> 6;                // S_/16 == 64
    const int q0   = qt << 4;
    const int b    = bh >> 4;                  // H_ == 16
    const int h    = bh & (H_ - 1);

    const _Float16* Q  = qb  + (size_t)bh * S_ * D_;
    const _Float16* Kk = kb  + (size_t)bh * S_ * D_;
    const _Float16* Vt = vtb + (size_t)bh * D_ * S_;

    const v16h aq0 = ld_frag_a(Q + q0 * D_ +  0, D_, lane);
    const v16h aq1 = ld_frag_a(Q + q0 * D_ + 32, D_, lane);

    float m_r[8], l_r[8];
#pragma unroll
    for (int r = 0; r < 8; ++r) { m_r[r] = -3.0e38f; l_r[r] = 0.0f; }
    v8f o0 = {}, o1 = {}, o2 = {}, o3 = {};

    const int row = (lane >> 4) << 3;
    const int col = lane & 15;
    _Float16* myp = lds_p[wid];

    for (int kk = 0; kk < q0 + 16; kk += 32) {
        // scores: two 16x16 tiles (keys kk..+15, kk+16..+31), K-dim = D = 64
        v8f s0 = {}, s1 = {};
        s0 = wmma_f16(aq0, ld_frag_b(Kk + (kk     ) * D_ +  0, D_, lane), s0);
        s0 = wmma_f16(aq1, ld_frag_b(Kk + (kk     ) * D_ + 32, D_, lane), s0);
        s1 = wmma_f16(aq0, ld_frag_b(Kk + (kk + 16) * D_ +  0, D_, lane), s1);
        s1 = wmma_f16(aq1, ld_frag_b(Kk + (kk + 16) * D_ + 32, D_, lane), s1);

#pragma unroll
        for (int r = 0; r < 8; ++r) {
            const int q = q0 + row + r;
            float v0 = s0[r] * 0.125f;               // 1/sqrt(64)
            float v1 = s1[r] * 0.125f;
            if (kk + col > q)      v0 = -10000.0f;   // causal MASKED_BIAS
            if (kk + 16 + col > q) v1 = -10000.0f;
            float mx = fmaxf(v0, v1);
#pragma unroll
            for (int msk = 1; msk < 16; msk <<= 1)
                mx = fmaxf(mx, __shfl_xor(mx, msk, 32));
            const float mnew = fmaxf(m_r[r], mx);
            const float corr = __expf(m_r[r] - mnew);
            const float p0   = __expf(v0 - mnew);
            const float p1   = __expf(v1 - mnew);
            float rs = p0 + p1;
#pragma unroll
            for (int msk = 1; msk < 16; msk <<= 1)
                rs += __shfl_xor(rs, msk, 32);
            l_r[r] = l_r[r] * corr + rs;
            m_r[r] = mnew;
            o0[r] *= corr; o1[r] *= corr; o2[r] *= corr; o3[r] *= corr;
            myp[(row + r) * 32 + col]      = (_Float16)p0;  // C-layout -> LDS
            myp[(row + r) * 32 + 16 + col] = (_Float16)p1;
        }
        asm volatile("s_wait_dscnt 0" ::: "memory");
        const v16h pa = ld_frag_a(myp, 32, lane);    // LDS -> A-layout
        o0 = wmma_f16(pa, ld_frag_b(Vt +  0 * S_ + kk, S_, lane), o0);
        o1 = wmma_f16(pa, ld_frag_b(Vt + 16 * S_ + kk, S_, lane), o1);
        o2 = wmma_f16(pa, ld_frag_b(Vt + 32 * S_ + kk, S_, lane), o2);
        o3 = wmma_f16(pa, ld_frag_b(Vt + 48 * S_ + kk, S_, lane), o3);
    }

#pragma unroll
    for (int r = 0; r < 8; ++r) {
        const float inv = 1.0f / l_r[r];
        o0[r] *= inv; o1[r] *= inv; o2[r] *= inv; o3[r] *= inv;
    }
    // merged heads: [B*S, E]
#pragma unroll
    for (int r = 0; r < 8; ++r) {
        const size_t base = ((size_t)(b * S_ + q0 + row + r)) * E_ + h * D_ + col;
        attn_out[base +  0] = (_Float16)o0[r];
        attn_out[base + 16] = (_Float16)o1[r];
        attn_out[base + 32] = (_Float16)o2[r];
        attn_out[base + 48] = (_Float16)o3[r];
    }
}

// ---- Output projection: [B*S,E] x [E,E]^T + bias -> f32 out ----------------
// Wave tile 32x64, same B-fragment sharing as gemm_qkv.
__global__ void __launch_bounds__(256)
gemm_proj(const _Float16* __restrict__ A, const _Float16* __restrict__ W,
          const float* __restrict__ bias, float* __restrict__ out) {
    const int lane = threadIdx.x & 31;
    const int gw   = blockIdx.x * 8 + (threadIdx.x >> 5);
    const int TN   = E_ / 64;                    // 16 n-tiles
    const int m0   = (gw / TN) << 5;             // 32-row tiles
    const int n0   = (gw % TN) << 6;

    v8f accLo[4] = {}, accHi[4] = {};
    const _Float16* alo = A + (size_t)m0 * E_;
    const _Float16* ahi = A + (size_t)(m0 + 16) * E_;
    for (int k0 = 0; k0 < E_; k0 += 32) {
        __builtin_prefetch(alo + k0 + 256, 0, 0);
        const v16h aLo = ld_frag_a(alo + k0, E_, lane);
        const v16h aHi = ld_frag_a(ahi + k0, E_, lane);
#pragma unroll
        for (int t = 0; t < 4; ++t) {
            const v16h bf = ld_frag_b(W + (size_t)(n0 + (t << 4)) * E_ + k0,
                                      E_, lane);
            accLo[t] = wmma_f16(aLo, bf, accLo[t]);
            accHi[t] = wmma_f16(aHi, bf, accHi[t]);
        }
    }
    const int row = (lane >> 4) << 3;
    const int col = lane & 15;
#pragma unroll
    for (int t = 0; t < 4; ++t) {
        const int oc   = n0 + (t << 4) + col;
        const float bv = bias[oc];
#pragma unroll
        for (int r = 0; r < 8; ++r) {
            out[(size_t)(m0 + row + r) * E_ + oc]      = accLo[t][r] + bv;
            out[(size_t)(m0 + 16 + row + r) * E_ + oc] = accHi[t][r] + bv;
        }
    }
}

// ---------------------------------------------------------------------------
extern "C" void kernel_launch(void* const* d_in, const int* in_sizes, int n_in,
                              void* d_out, int out_size, void* d_ws, size_t ws_size,
                              hipStream_t stream) {
    const float* hidden   = (const float*)d_in[0];
    const float* aw_sub   = (const float*)d_in[1];
    const float* aw_wl    = (const float*)d_in[2];
    const float* aw_W     = (const float*)d_in[3];
    const float* aw_B     = (const float*)d_in[4];
    const float* ab_sub   = (const float*)d_in[5];
    const float* ab_wl    = (const float*)d_in[6];
    const float* ab_W     = (const float*)d_in[7];
    const float* ab_B     = (const float*)d_in[8];
    const float* pw_sub   = (const float*)d_in[9];
    const float* pw_wl    = (const float*)d_in[10];
    const float* pw_W     = (const float*)d_in[11];
    const float* pw_B     = (const float*)d_in[12];
    const float* pb_sub   = (const float*)d_in[13];
    const float* pb_wl    = (const float*)d_in[14];
    const float* pb_W     = (const float*)d_in[15];
    const float* pb_B     = (const float*)d_in[16];

    // workspace carve-up (256B aligned)
    char*  ws  = (char*)d_ws;
    size_t off = 0;
    auto take = [&](size_t bytes) -> char* {
        char* p = ws + off;
        off = (off + bytes + 255) & ~(size_t)255;
        return p;
    };
    _Float16* x16    = (_Float16*)take((size_t)B_ * S_ * E_ * 2);       // 16 MB
    _Float16* wqkv16 = (_Float16*)take((size_t)3 * E_ * E_ * 2);        //  6 MB
    float*    bqkv   = (float*)   take((size_t)3 * E_ * 4);
    _Float16* wp16   = (_Float16*)take((size_t)E_ * E_ * 2);            //  2 MB
    float*    bp     = (float*)   take((size_t)E_ * 4);
    _Float16* qbuf   = (_Float16*)take((size_t)B_ * H_ * S_ * D_ * 2);  // 16 MB
    _Float16* kbuf   = (_Float16*)take((size_t)B_ * H_ * S_ * D_ * 2);  // 16 MB
    _Float16* vtbuf  = (_Float16*)take((size_t)B_ * H_ * D_ * S_ * 2);  // 16 MB
    _Float16* attn16 = (_Float16*)take((size_t)B_ * S_ * E_ * 2);       // 16 MB
    (void)ws_size; (void)n_in; (void)in_sizes; (void)out_size;

    // 1) generate weights/biases (f16 weights, f32 biases)
    gen_weight_f16<<<(3 * E_ * E_) / 256, 256, 0, stream>>>(
        aw_sub, aw_wl, aw_W, aw_B, wqkv16, 3 * E_ * E_);
    gen_weight_f16<<<(E_ * E_) / 256, 256, 0, stream>>>(
        pw_sub, pw_wl, pw_W, pw_B, wp16, E_ * E_);
    gen_bias_f32<<<(3 * E_) / 256, 256, 0, stream>>>(
        ab_sub, ab_wl, ab_W, ab_B, bqkv, 3 * E_);
    gen_bias_f32<<<E_ / 256, 256, 0, stream>>>(
        pb_sub, pb_wl, pb_W, pb_B, bp, E_);

    // 2) cast hidden states to f16
    cast_to_f16<<<((size_t)B_ * S_ * E_ / 4) / 256, 256, 0, stream>>>(hidden, x16);

    // 3) QKV GEMM (8192x3072x1024), wave-tile 32x64 -> 256*48 wave tiles
    gemm_qkv<<<(256 * 48) / 8, 256, 0, stream>>>(x16, wqkv16, bqkv,
                                                 qbuf, kbuf, vtbuf);

    // 4) causal flash attention, one wave per 16-query tile
    attn_kernel<<<(B_ * H_ * (S_ / 16)) / 8, 256, 0, stream>>>(
        qbuf, kbuf, vtbuf, attn16);

    // 5) output projection (8192x1024x1024), wave-tile 32x64 -> fp32 d_out
    gemm_proj<<<(256 * 16) / 8, 256, 0, stream>>>(attn16, wp16, bp,
                                                  (float*)d_out);
}
